// GINLSTM_2LayerDropout_24507083391722
// MI455X (gfx1250) — matrix operand order repaired and checked by reading, not compiled, not run on hardware
//
#include <hip/hip_runtime.h>

typedef __attribute__((ext_vector_type(16))) __bf16 v16bf;
typedef __attribute__((ext_vector_type(8)))  float  v8f;

#define T_STEPS 128
#define N_NODES 1000
#define E_EDGES 16000
#define F_DIM   397
#define F_PAD   416   // 13*32, rows 16B-aligned (416*4 = 1664)
#define H_DIM   128
#define D_DIM   8
#define KC1     13    // 416/32
#define KC2     4     // 128/32
#define M_TILES 63    // ceil(1000/16)

__device__ __forceinline__ float sigmoidf(float x) { return 1.0f / (1.0f + __expf(-x)); }

__device__ __forceinline__ void put4(v16bf& a, int base, float4 p) {
  a[base + 0] = (__bf16)p.x;
  a[base + 1] = (__bf16)p.y;
  a[base + 2] = (__bf16)p.z;
  a[base + 3] = (__bf16)p.w;
}

// ---------------- utility kernels ----------------
// s1p[t,n,k] = (k < F) ? x[t,n,k] : 0   (padded row stride F_PAD)
__global__ void k_init_s1(const float* __restrict__ x, float* __restrict__ s1p, long n_pad) {
  long i = (long)blockIdx.x * blockDim.x + threadIdx.x;
  long stride = (long)gridDim.x * blockDim.x;
  for (; i < n_pad; i += stride) {
    long row = i / F_PAD;
    int  k   = (int)(i - row * F_PAD);
    s1p[i] = (k < F_DIM) ? x[row * F_DIM + k] : 0.0f;
  }
}

__global__ void k_zero_f32(float* __restrict__ dst, long n) {
  long i = (long)blockIdx.x * blockDim.x + threadIdx.x;
  if (i < n) dst[i] = 0.0f;
}

// Pack weight w[H x K] (row-major) into WMMA B-fragment layout (bf16):
// element index = ((kc*8 + nt)*32 + lane)*16 + e
// h = lane>>4 : e<8 -> K = kc*32 + 8h + e ; e>=8 -> K = kc*32 + 16 + 8h + (e-8) ; col = nt*16 + (lane&15)
__global__ void k_pack_b(const float* __restrict__ w, __bf16* __restrict__ dst, int K, int total) {
  int i = blockIdx.x * blockDim.x + threadIdx.x;
  if (i >= total) return;
  int e    = i & 15;
  int lane = (i >> 4) & 31;
  int nt   = (i >> 9) & 7;
  int kc   = i >> 12;
  int col  = nt * 16 + (lane & 15);
  int k    = kc * 32 + ((lane >> 4) << 3) + (e & 7) + ((e >> 3) << 4);
  dst[i] = (__bf16)(k < K ? w[col * K + k] : 0.0f);
}

// Scatter-add: one block per (t, edge); s[t,dst,:] += xin[t,src,:]
// F = logical feature count, FS_in / FS_out = row strides of xin / s
__global__ void k_scatter_add(const float* __restrict__ xin, float* __restrict__ s,
                              const int* __restrict__ ei, int F, int FS_in, int FS_out) {
  int b = blockIdx.x;
  int t = b / E_EDGES;
  int e = b - t * E_EDGES;
  const int* eit = ei + (long)t * 2 * E_EDGES;
  int src = eit[e];
  int dst = eit[E_EDGES + e];
  const float* xr = xin + ((long)t * N_NODES + src) * FS_in;
  float* sr = s + ((long)t * N_NODES + dst) * FS_out;
  for (int f = threadIdx.x; f < F; f += blockDim.x)
    atomicAdd(&sr[f], xr[f]);
}

// ---------------- GIN GEMM layer 1: h1 = relu(s1p @ w1^T + b1) ----------------
// grid = (T, 8), block = 256 (8 waves); wave handles one 16-row node tile, all 8 H-tiles.
__global__ void __launch_bounds__(256) k_gin_gemm1(
    const float* __restrict__ s1p, const __bf16* __restrict__ pB,
    const float* __restrict__ b1, float* __restrict__ h1) {
  int t    = blockIdx.x;
  int wave = threadIdx.x >> 5;
  int lane = threadIdx.x & 31;
  int mtile = blockIdx.y * 8 + wave;
  if (mtile >= M_TILES) return;               // wave-uniform exit; EXEC all-1 for WMMA
  int rlo  = lane & 15;
  int half = lane >> 4;
  int arow = mtile * 16 + rlo;
  int rclamp = arow < N_NODES ? arow : 0;     // safe pointer; bad rows never stored
  const float* arp = s1p + ((long)t * N_NODES + rclamp) * F_PAD;

  v8f acc[8];
#pragma unroll
  for (int n = 0; n < 8; n++)
#pragma unroll
    for (int r = 0; r < 8; r++) acc[n][r] = 0.0f;

  for (int kc = 0; kc < KC1; kc++) {
    int kbase = kc * 32 + half * 8;
    __builtin_prefetch(arp + kbase + 32, 0, 1);
    float4 p0 = *(const float4*)(arp + kbase);
    float4 p1 = *(const float4*)(arp + kbase + 4);
    float4 p2 = *(const float4*)(arp + kbase + 16);
    float4 p3 = *(const float4*)(arp + kbase + 20);
    v16bf a;
    put4(a, 0, p0);  put4(a, 4, p1);
    put4(a, 8, p2);  put4(a, 12, p3);
    const v16bf* bp = (const v16bf*)(pB + ((long)(kc * 8) * 32 + lane) * 16);
#pragma unroll
    for (int nt = 0; nt < 8; nt++) {
      v16bf b = bp[nt * 32];
      acc[nt] = __builtin_amdgcn_wmma_f32_16x16x32_bf16(
          false, a, false, b, (short)0, acc[nt], false, false);
    }
  }

#pragma unroll
  for (int nt = 0; nt < 8; nt++) {
    int col = nt * 16 + rlo;
    float bias = b1[col];
#pragma unroll
    for (int r = 0; r < 8; r++) {
      int m = mtile * 16 + half * 8 + r;
      if (m < N_NODES) {
        float v = acc[nt][r] + bias;
        h1[((long)t * N_NODES + m) * H_DIM + col] = v > 0.0f ? v : 0.0f;
      }
    }
  }
}

// ---------------- GIN GEMM layer 2 + fused mean pool ----------------
__global__ void __launch_bounds__(256) k_gin_gemm2(
    const float* __restrict__ s2, const __bf16* __restrict__ pB,
    const float* __restrict__ b2, float* __restrict__ pooled) {
  int t    = blockIdx.x;
  int wave = threadIdx.x >> 5;
  int lane = threadIdx.x & 31;
  int mtile = blockIdx.y * 8 + wave;
  if (mtile >= M_TILES) return;
  int rlo  = lane & 15;
  int half = lane >> 4;
  int arow = mtile * 16 + rlo;
  int rclamp = arow < N_NODES ? arow : 0;
  const float* arp = s2 + ((long)t * N_NODES + rclamp) * H_DIM;

  v8f acc[8];
#pragma unroll
  for (int n = 0; n < 8; n++)
#pragma unroll
    for (int r = 0; r < 8; r++) acc[n][r] = 0.0f;

#pragma unroll
  for (int kc = 0; kc < KC2; kc++) {
    int kbase = kc * 32 + half * 8;
    float4 p0 = *(const float4*)(arp + kbase);
    float4 p1 = *(const float4*)(arp + kbase + 4);
    float4 p2 = *(const float4*)(arp + kbase + 16);
    float4 p3 = *(const float4*)(arp + kbase + 20);
    v16bf a;
    put4(a, 0, p0);  put4(a, 4, p1);
    put4(a, 8, p2);  put4(a, 12, p3);
    const v16bf* bp = (const v16bf*)(pB + ((long)(kc * 8) * 32 + lane) * 16);
#pragma unroll
    for (int nt = 0; nt < 8; nt++) {
      v16bf b = bp[nt * 32];
      acc[nt] = __builtin_amdgcn_wmma_f32_16x16x32_bf16(
          false, a, false, b, (short)0, acc[nt], false, false);
    }
  }

  const float inv_n = 1.0f / (float)N_NODES;
#pragma unroll
  for (int nt = 0; nt < 8; nt++) {
    int col = nt * 16 + rlo;
    float bias = b2[col];
    float sum = 0.0f;
#pragma unroll
    for (int r = 0; r < 8; r++) {
      int m = mtile * 16 + half * 8 + r;
      if (m < N_NODES) {
        float v = acc[nt][r] + bias;
        sum += (v > 0.0f ? v : 0.0f);
      }
    }
    atomicAdd(&pooled[t * H_DIM + col], sum * inv_n);
  }
}

// ---------------- LSTM scan + FC (sequential, single block) ----------------
__global__ void __launch_bounds__(512) k_lstm(
    const float* __restrict__ pooled, const float* __restrict__ emb,
    const int* __restrict__ dok,
    const float* __restrict__ w_ih, const float* __restrict__ w_hh,
    const float* __restrict__ b_ih, const float* __restrict__ b_hh,
    const float* __restrict__ w_fc, const float* __restrict__ b_fc,
    float* __restrict__ out) {
  __shared__ float sh_seq[H_DIM + D_DIM];
  __shared__ float sh_h[H_DIM];
  __shared__ float sh_c[H_DIM];
  __shared__ float sh_g[4 * H_DIM];
  __shared__ float sh_red[H_DIM];
  int tid = threadIdx.x;
  if (tid < H_DIM) { sh_h[tid] = 0.0f; sh_c[tid] = 0.0f; }
  __syncthreads();

  for (int t = 0; t < T_STEPS; t++) {
    if (tid < H_DIM) sh_seq[tid] = pooled[t * H_DIM + tid];
    if (tid < D_DIM) sh_seq[H_DIM + tid] = emb[dok[t] * D_DIM + tid];
    __syncthreads();

    float acc = b_ih[tid] + b_hh[tid];
    const float* wi = w_ih + (long)tid * (H_DIM + D_DIM);
    for (int k = 0; k < H_DIM + D_DIM; k++) acc += wi[k] * sh_seq[k];
    const float* wh = w_hh + (long)tid * H_DIM;
    for (int k = 0; k < H_DIM; k++) acc += wh[k] * sh_h[k];
    sh_g[tid] = acc;
    __syncthreads();

    if (tid < H_DIM) {
      float gi = sigmoidf(sh_g[tid]);
      float gf = sigmoidf(sh_g[H_DIM + tid]);
      float gg = tanhf(sh_g[2 * H_DIM + tid]);
      float go = sigmoidf(sh_g[3 * H_DIM + tid]);
      float c = gf * sh_c[tid] + gi * gg;
      float h = go * tanhf(c);
      sh_c[tid] = c;
      sh_h[tid] = h;
      sh_red[tid] = h * w_fc[tid];
    }
    __syncthreads();
    for (int s = H_DIM / 2; s > 0; s >>= 1) {
      if (tid < s) sh_red[tid] += sh_red[tid + s];
      __syncthreads();
    }
    if (tid == 0) out[t] = sh_red[0] + b_fc[0];
    __syncthreads();
  }
}

extern "C" void kernel_launch(void* const* d_in, const int* in_sizes, int n_in,
                              void* d_out, int out_size, void* d_ws, size_t ws_size,
                              hipStream_t stream) {
  const float* x    = (const float*)d_in[0];
  const int*   ei   = (const int*)d_in[1];
  const int*   dok  = (const int*)d_in[2];
  const float* w1   = (const float*)d_in[3];
  const float* b1   = (const float*)d_in[4];
  const float* w2   = (const float*)d_in[5];
  const float* b2   = (const float*)d_in[6];
  const float* emb  = (const float*)d_in[7];
  const float* w_ih = (const float*)d_in[8];
  const float* w_hh = (const float*)d_in[9];
  const float* b_ih = (const float*)d_in[10];
  const float* b_hh = (const float*)d_in[11];
  const float* w_fc = (const float*)d_in[12];
  const float* b_fc = (const float*)d_in[13];
  float* out = (float*)d_out;

  char* ws = (char*)d_ws;
  const size_t S1_BYTES = (size_t)T_STEPS * N_NODES * F_PAD * 4;  // 212,992,000
  const size_t H1_BYTES = (size_t)T_STEPS * N_NODES * H_DIM * 4;  //  65,536,000
  float*  s1p    = (float*)ws;                       // [T,N,F_PAD]
  float*  s2     = (float*)ws;                       // [T,N,H] aliases s1p (dead after gemm1)
  float*  h1     = (float*)(ws + S1_BYTES);          // [T,N,H]
  float*  pooled = (float*)(ws + S1_BYTES + H1_BYTES);   // [T,H]
  __bf16* pB1    = (__bf16*)(ws + S1_BYTES + H1_BYTES + 65536);
  __bf16* pB2    = pB1 + (size_t)KC1 * 8 * 32 * 16;

  // pack weights into WMMA B-fragment layout (bf16)
  { int tot = KC1 * 8 * 32 * 16;
    k_pack_b<<<(tot + 255) / 256, 256, 0, stream>>>(w1, pB1, F_DIM, tot); }
  { int tot = KC2 * 8 * 32 * 16;
    k_pack_b<<<(tot + 255) / 256, 256, 0, stream>>>(w2, pB2, H_DIM, tot); }

  // s1p = pad(x) ; pooled = 0
  { long n = (long)T_STEPS * N_NODES * F_PAD;
    k_init_s1<<<4096, 256, 0, stream>>>(x, s1p, n); }
  k_zero_f32<<<(T_STEPS * H_DIM + 255) / 256, 256, 0, stream>>>(pooled, T_STEPS * H_DIM);

  // layer 1: scatter + WMMA GEMM
  k_scatter_add<<<T_STEPS * E_EDGES, 128, 0, stream>>>(x, s1p, ei, F_DIM, F_DIM, F_PAD);
  k_gin_gemm1<<<dim3(T_STEPS, 8), 256, 0, stream>>>(s1p, pB1, b1, h1);

  // layer 2: s2 = h1, scatter, WMMA GEMM + fused mean pool
  hipMemcpyAsync(s2, h1, H1_BYTES, hipMemcpyDeviceToDevice, stream);
  k_scatter_add<<<T_STEPS * E_EDGES, 128, 0, stream>>>(h1, s2, ei, H_DIM, H_DIM, H_DIM);
  k_gin_gemm2<<<dim3(T_STEPS, 8), 256, 0, stream>>>(s2, pB2, b2, pooled);

  // sequential LSTM + FC
  k_lstm<<<1, 512, 0, stream>>>(pooled, emb, dok, w_ih, w_hh, b_ih, b_hh,
                                w_fc, b_fc, out);
}
